// MultiHeadAttenionBlock_13365938225284
// MI455X (gfx1250) — matrix operand order repaired and checked by reading, not compiled, not run on hardware
//
#include <hip/hip_runtime.h>
#include <hip/hip_bf16.h>
#include <stdint.h>

// ---------------------------------------------------------------------------
// Transformer block (pre-LN MHA + 3-layer FF) for MI455X / gfx1250 (wave32).
// All heavy math via v_wmma_f32_16x16x32_bf16; GEMM staging via CDNA5 async
// global->LDS copies (ASYNCcnt); flash-attention with WMMA for QK^T and PV.
// Shapes fixed: B=4, T=2048, C=1024, H=16, D=64, FF=4096, M=B*T=8192.
//
// GEMM geometry: block tile 128(M) x 256(N), K-step 32, 8 waves, each wave
// owns a 64x64 tile -> 16 WMMAs per K-step vs 8 fragment loads (16 ds_b128):
// maximizes WMMA issue density per LDS byte and per barrier.
// ---------------------------------------------------------------------------

typedef __bf16 bf16_t;
typedef __attribute__((ext_vector_type(16))) __bf16 v16bf;
typedef __attribute__((ext_vector_type(8)))  __bf16 v8bf;
typedef __attribute__((ext_vector_type(8)))  float   v8f;

#define B_DIM  4
#define T_DIM  2048
#define C_DIM  1024
#define H_DIM  16
#define D_DIM  64
#define FF_DIM 4096
#define M_DIM  (B_DIM * T_DIM)

static __device__ __forceinline__ v16bf cat8(v8bf lo, v8bf hi) {
  return __builtin_shufflevector(lo, hi, 0,1,2,3,4,5,6,7,8,9,10,11,12,13,14,15);
}
static __device__ __forceinline__ v8f wmma_bf16(v16bf a, v16bf b, v8f c) {
  // D = A(16x32 bf16) * B(32x16 bf16) + C(16x16 f32)
  return __builtin_amdgcn_wmma_f32_16x16x32_bf16(false, a, false, b, (short)0, c,
                                                 false, false);
}
static __device__ __forceinline__ uint32_t lds_off(const void* p) {
  // generic (shared) pointer low 32 bits == LDS byte offset
  return (uint32_t)(uintptr_t)p;
}
static __device__ __forceinline__ void async_copy16(uint32_t lds, const void* g) {
  // CDNA5 async global->LDS, 16B per lane, tracked by ASYNCcnt
  asm volatile("global_load_async_to_lds_b128 %0, %1, off" :: "v"(lds), "v"(g) : "memory");
}

#define LPAD 40  // LDS row stride in bf16 elems (32 data + 8 pad), 80B (16B-mult)

// ---------------------------------------------------------------------------
// Generic bf16 WMMA GEMM: out = A[M,K] * Bt[N,K]^T (+bias)(+res)(relu?)
// Double-buffered async LDS staging. Optional f32 and/or bf16 outputs;
// bfmode==1 writes the bf16 output transposed per-(b,h) as V^T[b,h,d,t].
// ---------------------------------------------------------------------------
__global__ __launch_bounds__(256) void gemm_bf16_wmma(
    const bf16_t* __restrict__ A, const bf16_t* __restrict__ Bt,
    const float* __restrict__ bias, const float* __restrict__ res,
    float* __restrict__ outF, bf16_t* __restrict__ outB,
    int M, int N, int K, int relu, int bfmode)
{
  __shared__ bf16_t sA[2][128 * LPAD];   // 2 x 10240 B
  __shared__ bf16_t sB[2][256 * LPAD];   // 2 x 20480 B   (total 60KB)

  const int tid   = threadIdx.x;
  const int rowM0 = blockIdx.y * 128;
  const int colN0 = blockIdx.x * 256;
  const int nK    = K >> 5;

  auto issue = [&](int buf, int kb) {
    const int k0 = kb << 5;
    #pragma unroll
    for (int i = 0; i < 6; ++i) {        // 1536 16B-chunks: 512 A + 1024 B
      const int c = tid + i * 256;
      if (c < 512) {
        const int row = c >> 2, seg = c & 3;
        async_copy16(lds_off(&sA[buf][row * LPAD + seg * 8]),
                     A + (size_t)(rowM0 + row) * K + k0 + seg * 8);
      } else {
        const int cb = c - 512;
        const int row = cb >> 2, seg = cb & 3;
        async_copy16(lds_off(&sB[buf][row * LPAD + seg * 8]),
                     Bt + (size_t)(colN0 + row) * K + k0 + seg * 8);
      }
    }
  };

  const int lane = tid & 31, wid = tid >> 5;
  const int wm = (wid & 1) * 64;   // 2 waves span M=128
  const int wn = (wid >> 1) * 64;  // 4 waves span N=256
  const int lm = lane & 15, sel = lane >> 4;

  const v8f vzero = {0.f,0.f,0.f,0.f,0.f,0.f,0.f,0.f};
  v8f acc[4][4];
  #pragma unroll
  for (int mi = 0; mi < 4; ++mi)
    #pragma unroll
    for (int ni = 0; ni < 4; ++ni) acc[mi][ni] = vzero;

  issue(0, 0);
  for (int kb = 0; kb < nK; ++kb) {
    const int cur = kb & 1;
    if (kb + 1 < nK) {
      issue(cur ^ 1, kb + 1);
      // per-wave ASYNCcnt completes in order: <=6 left ==> current tile done
      asm volatile("s_wait_asynccnt 0x6" ::: "memory");
    } else {
      asm volatile("s_wait_asynccnt 0x0" ::: "memory");
    }
    __syncthreads();

    v16bf af[4], bfr[4];
    #pragma unroll
    for (int mi = 0; mi < 4; ++mi) {   // A frag: lane row lm, K {sel*8..+8, 16+sel*8..+8}
      const bf16_t* ap = &sA[cur][(wm + mi * 16 + lm) * LPAD];
      af[mi] = cat8(*(const v8bf*)(ap + sel * 8),
                    *(const v8bf*)(ap + 16 + sel * 8));
    }
    #pragma unroll
    for (int ni = 0; ni < 4; ++ni) {   // B frag: lane col lm, K {sel*8..+8, 16+sel*8..+8}
      const bf16_t* bp = &sB[cur][(wn + ni * 16 + lm) * LPAD];
      bfr[ni] = cat8(*(const v8bf*)(bp + sel * 8),
                     *(const v8bf*)(bp + 16 + sel * 8));
    }
    #pragma unroll
    for (int mi = 0; mi < 4; ++mi)
      #pragma unroll
      for (int ni = 0; ni < 4; ++ni)
        acc[mi][ni] = wmma_bf16(af[mi], bfr[ni], acc[mi][ni]);
    __syncthreads();   // protect buffer before it is refilled
  }

  #pragma unroll
  for (int mi = 0; mi < 4; ++mi) {
    #pragma unroll
    for (int ni = 0; ni < 4; ++ni) {
      const int col = colN0 + wn + ni * 16 + lm;
      const float bb = bias ? bias[col] : 0.0f;
      #pragma unroll
      for (int r = 0; r < 8; ++r) {
        const int row = rowM0 + wm + mi * 16 + sel * 8 + r; // C/D layout
        float v = acc[mi][ni][r] + bb;
        if (res)  v += res[(size_t)row * N + col];
        if (relu) v = fmaxf(v, 0.0f);
        if (outF) outF[(size_t)row * N + col] = v;
        if (outB) {
          size_t idx;
          if (bfmode == 0) {
            idx = (size_t)row * N + col;
          } else {            // V^T layout: [b, h, d, t]
            const int b = row >> 11, s = row & (T_DIM - 1);
            const int h = col >> 6,  d = col & 63;
            idx = ((size_t)((b * H_DIM + h) * D_DIM + d) << 11) + s;
          }
          outB[idx] = (bf16_t)v;
        }
      }
    }
  }
}

// ---------------------------------------------------------------------------
// Flash attention (causal). Grid (B*H, T/64), 128 threads = 4 waves, each
// wave owns 16 query rows. QK^T and PV via WMMA; softmax stats kept in the
// C/D fragment layout; P goes through a per-wave LDS patch to A-layout.
// ---------------------------------------------------------------------------
__global__ __launch_bounds__(128) void flash_attn_wmma(
    const bf16_t* __restrict__ Q, const bf16_t* __restrict__ Kb,
    const bf16_t* __restrict__ Vt, bf16_t* __restrict__ O)
{
  __shared__ bf16_t sP[4][16 * LPAD];

  const int bh = blockIdx.x;                 // b*H + h
  const int h  = bh & (H_DIM - 1);
  const int wid = threadIdx.x >> 5, lane = threadIdx.x & 31;
  const int lm = lane & 15, sel = lane >> 4;
  const int qbase = blockIdx.y * 64 + wid * 16;
  const size_t rowBase = (size_t)(bh >> 4) * T_DIM;   // b*T
  const float scale = 0.03125f;                       // C^-0.5 = 1/32

  // Q fragments for both 32-wide d-slices (A-operand, loaded once)
  const bf16_t* qrow = Q + (rowBase + qbase + lm) * C_DIM + h * D_DIM;
  v16bf qa[2];
  #pragma unroll
  for (int dk = 0; dk < 2; ++dk)
    qa[dk] = cat8(*(const v8bf*)(qrow + dk * 32 + sel * 8),
                  *(const v8bf*)(qrow + dk * 32 + 16 + sel * 8));

  const v8f vzero = {0.f,0.f,0.f,0.f,0.f,0.f,0.f,0.f};
  v8f o[4]; float mrow[8], lrow[8];
  #pragma unroll
  for (int j = 0; j < 4; ++j) o[j] = vzero;
  #pragma unroll
  for (int r = 0; r < 8; ++r) { mrow[r] = -1e30f; lrow[r] = 0.0f; }

  for (int s0 = 0; s0 <= qbase + 15; s0 += 32) {
    // ---- S = Q K^T (16x32 tile of scores) ----
    v8f sacc[2] = {vzero, vzero};
    #pragma unroll
    for (int dk = 0; dk < 2; ++dk) {
      #pragma unroll
      for (int ni = 0; ni < 2; ++ni) {
        const bf16_t* krow =
            Kb + (rowBase + s0 + ni * 16 + lm) * C_DIM + h * D_DIM + dk * 32;
        sacc[ni] = wmma_bf16(qa[dk],
                             cat8(*(const v8bf*)(krow + sel * 8),
                                  *(const v8bf*)(krow + 16 + sel * 8)),
                             sacc[ni]);
      }
    }
    // ---- online softmax in fragment layout ----
    float p[2][8], corr[8];
    #pragma unroll
    for (int r = 0; r < 8; ++r) {
      const int t = qbase + sel * 8 + r;
      float rm = -1e30f;
      #pragma unroll
      for (int ni = 0; ni < 2; ++ni) {
        const int sk = s0 + ni * 16 + lm;
        float v = sacc[ni][r] * scale;
        if (sk > t) v = -1e30f;              // causal mask
        p[ni][r] = v;
        rm = fmaxf(rm, v);
      }
      #pragma unroll
      for (int msk = 8; msk >= 1; msk >>= 1) rm = fmaxf(rm, __shfl_xor(rm, msk, 32));
      const float mn = fmaxf(mrow[r], rm);
      corr[r] = __expf(mrow[r] - mn);
      mrow[r] = mn;
      float rs = 0.0f;
      #pragma unroll
      for (int ni = 0; ni < 2; ++ni) { p[ni][r] = __expf(p[ni][r] - mn); rs += p[ni][r]; }
      #pragma unroll
      for (int msk = 8; msk >= 1; msk >>= 1) rs += __shfl_xor(rs, msk, 32);
      lrow[r] = lrow[r] * corr[r] + rs;
    }
    #pragma unroll
    for (int j = 0; j < 4; ++j)
      #pragma unroll
      for (int r = 0; r < 8; ++r) o[j][r] *= corr[r];

    // ---- P: D-layout -> LDS -> A-layout (wave-synchronous) ----
    bf16_t* pl = &sP[wid][0];
    #pragma unroll
    for (int ni = 0; ni < 2; ++ni)
      #pragma unroll
      for (int r = 0; r < 8; ++r)
        pl[(sel * 8 + r) * LPAD + ni * 16 + lm] = (bf16_t)p[ni][r];
    const v16bf pa = cat8(*(const v8bf*)&pl[lm * LPAD + sel * 8],
                          *(const v8bf*)&pl[lm * LPAD + 16 + sel * 8]);

    // ---- O += P V (V pre-transposed: rows are d, contiguous in t) ----
    #pragma unroll
    for (int j = 0; j < 4; ++j) {
      const bf16_t* vr =
          Vt + ((size_t)bh * D_DIM + j * 16 + lm) * T_DIM + s0 + sel * 8;
      o[j] = wmma_bf16(pa, cat8(*(const v8bf*)vr, *(const v8bf*)(vr + 16)), o[j]);
    }
  }

  // normalize + write concat-heads bf16 output
  float linv[8];
  #pragma unroll
  for (int r = 0; r < 8; ++r) linv[r] = __frcp_rn(lrow[r]);
  #pragma unroll
  for (int j = 0; j < 4; ++j)
    #pragma unroll
    for (int r = 0; r < 8; ++r) {
      const int t = qbase + sel * 8 + r;
      const int d = j * 16 + lm;
      O[(rowBase + t) * C_DIM + h * D_DIM + d] = (bf16_t)(o[j][r] * linv[r]);
    }
}

// ---------------------------------------------------------------------------
// LayerNorm (fp32 in) -> bf16 out. One 256-thread block per row of C=1024.
// ---------------------------------------------------------------------------
__global__ __launch_bounds__(256) void layernorm_to_bf16(
    const float* __restrict__ x, const float* __restrict__ g,
    const float* __restrict__ bta, bf16_t* __restrict__ out)
{
  const int row = blockIdx.x;
  const float* xr = x + (size_t)row * C_DIM;
  float v[4], s = 0.f, s2 = 0.f;
  #pragma unroll
  for (int i = 0; i < 4; ++i) {
    v[i] = xr[threadIdx.x + i * 256];
    s += v[i]; s2 += v[i] * v[i];
  }
  #pragma unroll
  for (int m = 16; m >= 1; m >>= 1) {
    s  += __shfl_xor(s,  m, 32);
    s2 += __shfl_xor(s2, m, 32);
  }
  __shared__ float red[2][8];
  const int wid = threadIdx.x >> 5, lane = threadIdx.x & 31;
  if (lane == 0) { red[0][wid] = s; red[1][wid] = s2; }
  __syncthreads();
  s = 0.f; s2 = 0.f;
  #pragma unroll
  for (int i = 0; i < 8; ++i) { s += red[0][i]; s2 += red[1][i]; }
  const float mu  = s * (1.0f / C_DIM);
  const float var = s2 * (1.0f / C_DIM) - mu * mu;
  const float inv = rsqrtf(var + 1e-5f);
  bf16_t* orow = out + (size_t)row * C_DIM;
  #pragma unroll
  for (int i = 0; i < 4; ++i) {
    const int c = threadIdx.x + i * 256;
    orow[c] = (bf16_t)((v[i] - mu) * inv * g[c] + bta[c]);
  }
}

// ---------------------------------------------------------------------------
// Weight prep: fp32 -> bf16 with transpose into [N][K] (B-operand layout).
// ---------------------------------------------------------------------------
__global__ void convert_transpose_bf16(const float* __restrict__ W,
                                       bf16_t* __restrict__ Wt, int K, int N)
{
  const size_t i = (size_t)blockIdx.x * 256 + threadIdx.x;   // over N*K
  if (i >= (size_t)K * N) return;
  const int n = (int)(i / K), k = (int)(i % K);
  Wt[i] = (bf16_t)W[(size_t)k * N + n];
}

// Wq/Wk/Wv stacked [H][C][D] -> [H*D][C] bf16
__global__ void convert_qkv_weight(const float* __restrict__ W,
                                   bf16_t* __restrict__ Wt)
{
  const size_t i = (size_t)blockIdx.x * 256 + threadIdx.x;   // over C*C
  const int n = (int)(i >> 10), c = (int)(i & (C_DIM - 1));
  const int h = n >> 6, d = n & 63;
  Wt[i] = (bf16_t)W[((size_t)h * C_DIM + c) * D_DIM + d];
}

// ---------------------------------------------------------------------------
extern "C" void kernel_launch(void* const* d_in, const int* in_sizes, int n_in,
                              void* d_out, int out_size, void* d_ws, size_t ws_size,
                              hipStream_t stream)
{
  (void)in_sizes; (void)n_in; (void)out_size; (void)ws_size;
  const float* x     = (const float*)d_in[0];
  const float* Wq    = (const float*)d_in[1];
  const float* Wk    = (const float*)d_in[2];
  const float* Wv    = (const float*)d_in[3];
  const float* Wproj = (const float*)d_in[4];
  const float* bproj = (const float*)d_in[5];
  const float* W1    = (const float*)d_in[6];
  const float* b1    = (const float*)d_in[7];
  const float* W2    = (const float*)d_in[8];
  const float* b2    = (const float*)d_in[9];
  const float* W3    = (const float*)d_in[10];
  const float* b3    = (const float*)d_in[11];
  const float* ln1g  = (const float*)d_in[12];
  const float* ln1b  = (const float*)d_in[13];
  const float* ln2g  = (const float*)d_in[14];
  const float* ln2b  = (const float*)d_in[15];

  char* p = (char*)d_ws;
  auto alloc = [&](size_t bytes) {
    char* r = p; p += (bytes + 255) & ~(size_t)255; return r;
  };
  const size_t MC = (size_t)M_DIM * C_DIM;
  bf16_t* hb   = (bf16_t*)alloc(MC * 2);          // ln1 out; reused for ln2 out
  bf16_t* qb   = (bf16_t*)alloc(MC * 2);          // q  [M][C]
  bf16_t* kbuf = (bf16_t*)alloc(MC * 2);          // k  [M][C]
  bf16_t* vt   = (bf16_t*)alloc(MC * 2);          // v^T [b,h,d,t]
  bf16_t* attn = (bf16_t*)alloc(MC * 2);          // attn out [M][C]
  bf16_t* f1   = qb;                              // alias: 4*M*C == M*FF
  bf16_t* f2   = (bf16_t*)alloc((size_t)M_DIM * FF_DIM * 2);
  float*  y    = (float*)alloc(MC * 4);           // x + attn@Wproj (fp32 residual)
  bf16_t* wqT  = (bf16_t*)alloc((size_t)C_DIM * C_DIM * 2);
  bf16_t* wkT  = (bf16_t*)alloc((size_t)C_DIM * C_DIM * 2);
  bf16_t* wvT  = (bf16_t*)alloc((size_t)C_DIM * C_DIM * 2);
  bf16_t* wpT  = (bf16_t*)alloc((size_t)C_DIM * C_DIM * 2);
  bf16_t* w1T  = (bf16_t*)alloc((size_t)C_DIM * FF_DIM * 2);
  bf16_t* w2T  = (bf16_t*)alloc((size_t)FF_DIM * FF_DIM * 2);
  bf16_t* w3T  = (bf16_t*)alloc((size_t)FF_DIM * C_DIM * 2);

  // --- weight prep (bf16 + transpose; all weights then live in L2: ~56MiB) ---
  const int gCC = (C_DIM * C_DIM) / 256;
  convert_qkv_weight<<<gCC, 256, 0, stream>>>(Wq, wqT);
  convert_qkv_weight<<<gCC, 256, 0, stream>>>(Wk, wkT);
  convert_qkv_weight<<<gCC, 256, 0, stream>>>(Wv, wvT);
  convert_transpose_bf16<<<gCC, 256, 0, stream>>>(Wproj, wpT, C_DIM, C_DIM);
  convert_transpose_bf16<<<(C_DIM * FF_DIM) / 256, 256, 0, stream>>>(W1, w1T, C_DIM, FF_DIM);
  convert_transpose_bf16<<<(FF_DIM * FF_DIM) / 256, 256, 0, stream>>>(W2, w2T, FF_DIM, FF_DIM);
  convert_transpose_bf16<<<(FF_DIM * C_DIM) / 256, 256, 0, stream>>>(W3, w3T, FF_DIM, C_DIM);

  // --- attention path ---
  layernorm_to_bf16<<<M_DIM, 256, 0, stream>>>(x, ln1g, ln1b, hb);
  dim3 gC(C_DIM / 256, M_DIM / 128);     // (4, 64)
  gemm_bf16_wmma<<<gC, 256, 0, stream>>>(hb, wqT, nullptr, nullptr, nullptr, qb,
                                         M_DIM, C_DIM, C_DIM, 0, 0);
  gemm_bf16_wmma<<<gC, 256, 0, stream>>>(hb, wkT, nullptr, nullptr, nullptr, kbuf,
                                         M_DIM, C_DIM, C_DIM, 0, 0);
  gemm_bf16_wmma<<<gC, 256, 0, stream>>>(hb, wvT, nullptr, nullptr, nullptr, vt,
                                         M_DIM, C_DIM, C_DIM, 0, 1);   // V^T epilogue
  dim3 gA(B_DIM * H_DIM, T_DIM / 64);    // (64, 32)
  flash_attn_wmma<<<gA, 128, 0, stream>>>(qb, kbuf, vt, attn);
  gemm_bf16_wmma<<<gC, 256, 0, stream>>>(attn, wpT, bproj, x, y, nullptr,
                                         M_DIM, C_DIM, C_DIM, 0, 0);

  // --- feed-forward path ---
  layernorm_to_bf16<<<M_DIM, 256, 0, stream>>>(y, ln2g, ln2b, hb);
  dim3 gF(FF_DIM / 256, M_DIM / 128);    // (16, 64)
  gemm_bf16_wmma<<<gF, 256, 0, stream>>>(hb, w1T, b1, nullptr, nullptr, f1,
                                         M_DIM, FF_DIM, C_DIM, 1, 0);
  gemm_bf16_wmma<<<gF, 256, 0, stream>>>(f1, w2T, b2, nullptr, nullptr, f2,
                                         M_DIM, FF_DIM, FF_DIM, 1, 0);
  gemm_bf16_wmma<<<gC, 256, 0, stream>>>(f2, w3T, b3, y, (float*)d_out, nullptr,
                                         M_DIM, C_DIM, FF_DIM, 0, 0);
}
